// GraphTransformer_49074296324301
// MI455X (gfx1250) — matrix operand order repaired and checked by reading, not compiled, not run on hardware
//
#include <hip/hip_runtime.h>
#include <hip/hip_bf16.h>
#include <stdint.h>

#define N_NODES  50000
#define N_EDGES  800000
#define HIDDEN   128
#define HEADS    4
#define HEAD_DIM 32
#define LAYERS   2

typedef __attribute__((ext_vector_type(16))) __bf16 v16bf;
typedef __attribute__((ext_vector_type(8)))  __bf16 v8bf;
typedef __attribute__((ext_vector_type(8)))  float  v8f;

// LDS weight panel: 128 columns x 128 K, padded +8 bf16 per column so the 16
// column-lanes of a B-fragment read hit distinct banks (stride 272B = 68 dw).
#define BPAD   8
#define BSTRIDE (128 + BPAD)   // bf16 elements per staged column

// ---------- CDNA5 async global->LDS copy (ASYNCcnt path) ------------------
// GLOBAL_LOAD_ASYNC_TO_LDS_B128: per-lane 16B copy from global memory into LDS, no VGPR
// round-trip, tracked by ASYNCcnt (ISA 15.18.3 op 98).
__device__ __forceinline__ void async_g2l_b128(unsigned lds_off, const void* gsrc) {
    asm volatile("global_load_async_to_lds_b128 %0, %1, off"
                 :: "v"(lds_off), "v"(gsrc) : "memory");
}
__device__ __forceinline__ void wait_asynccnt0() {
    asm volatile("s_wait_asynccnt 0x0" ::: "memory");
}

// ---------- ordered-uint mapping for float atomicMax (handles negatives) ----
__device__ __forceinline__ unsigned f2ord(float f) {
    unsigned u = __float_as_uint(f);
    return (u & 0x80000000u) ? ~u : (u | 0x80000000u);
}
__device__ __forceinline__ float ord2f(unsigned u) {
    return (u & 0x80000000u) ? __uint_as_float(u & 0x7FFFFFFFu)
                             : __uint_as_float(~u);
}

// ---------- weight prep: f32 [K,M] -> bf16 [M,K] -------------------------
__global__ void convT_kernel(const float* __restrict__ W, __bf16* __restrict__ Wt,
                             int K, int M) {
    int i = blockIdx.x * blockDim.x + threadIdx.x;
    if (i < K * M) {
        int k = i / M, m = i % M;
        Wt[(size_t)m * K + k] = (__bf16)W[i];
    }
}

// ---------- GEMM: out[rows,M] = A[rows,K] @ W[K,M] + bias (+res, relu) ----
// Block = 8 waves; each wave owns one 16-row strip with ALL A fragments in
// registers, then sweeps column tiles against an LDS-staged weight panel
// (staged via CDNA5 async global->LDS copies).
template <int K>
__global__ __launch_bounds__(256)
void gemm_bf16_wmma(const float* __restrict__ A,
                    const __bf16* __restrict__ Wt,     // [M,K] bf16
                    const float* __restrict__ bias,    // [M]
                    const float* __restrict__ residual,// [rows,M] or null
                    float* __restrict__ out,           // [rows,M]
                    int rows, int M, int relu) {
    __shared__ __align__(32) __bf16 Bs[128 * BSTRIDE];   // 34816 B

    constexpr int NKF = K / 32;            // A fragments per strip (4 or 8)
    const int wv    = threadIdx.x >> 5;
    const int strip = blockIdx.x * 8 + wv;               // 16-row strip id
    const bool valid = strip * 16 < rows;                // wave-uniform
    const int lane = threadIdx.x & 31;
    const int half = lane >> 4;
    const int lr   = lane & 15;

    // ---- load all A fragments for this strip into registers ----
    v16bf afrag[NKF];
    if (valid) {
        const float* abase = A + (size_t)(strip * 16 + lr) * K;
#pragma unroll
        for (int f = 0; f < NKF; ++f) {
            const int kb = f * 32;
            // half 0 -> K [kb+0..7],[kb+16..23]; half 1 -> [8..15],[24..31]
            const float4* ap0 = reinterpret_cast<const float4*>(abase + kb + half * 8);
            float4 a0 = ap0[0];
            float4 a1 = ap0[1];
            const float4* ap1 = reinterpret_cast<const float4*>(abase + kb + 16 + half * 8);
            float4 a2 = ap1[0];
            float4 a3 = ap1[1];
            v16bf af;
            af[0] = (__bf16)a0.x;  af[1] = (__bf16)a0.y;
            af[2] = (__bf16)a0.z;  af[3] = (__bf16)a0.w;
            af[4] = (__bf16)a1.x;  af[5] = (__bf16)a1.y;
            af[6] = (__bf16)a1.z;  af[7] = (__bf16)a1.w;
            af[8] = (__bf16)a2.x;  af[9] = (__bf16)a2.y;
            af[10] = (__bf16)a2.z; af[11] = (__bf16)a2.w;
            af[12] = (__bf16)a3.x; af[13] = (__bf16)a3.y;
            af[14] = (__bf16)a3.z; af[15] = (__bf16)a3.w;
            afrag[f] = af;
        }
    }

    const int mblocks = M >> 7;            // 128-column panels
    const int cl = threadIdx.x >> 1;       // LDS fill: column per thread pair
    const int hk = threadIdx.x & 1;        //            K-half per thread
    const unsigned ldsbase =
        (unsigned)(uintptr_t)(Bs + cl * BSTRIDE + hk * 64);

    for (int mb = 0; mb < mblocks; ++mb) {
        v8f acc[8];
        if (valid) {
#pragma unroll
            for (int t = 0; t < 8; ++t) acc[t] = (v8f){};
        }

#pragma unroll
        for (int kblk = 0; kblk < K / 128; ++kblk) {
            __syncthreads();               // previous panel fully consumed
            {   // async stage of 128 cols x 128 K (bf16) into LDS:
                // 8 x 16B per thread, all in flight on ASYNCcnt.
                const __bf16* wsrc = Wt + (size_t)(mb * 128 + cl) * K
                                        + kblk * 128 + hk * 64;
#pragma unroll
                for (int i = 0; i < 8; ++i)
                    async_g2l_b128(ldsbase + i * 16, wsrc + i * 8);
                wait_asynccnt0();
            }
            __syncthreads();

            if (valid) {
#pragma unroll
                for (int t = 0; t < 8; ++t) {          // 8 column tiles
                    const __bf16* bcol = Bs + (t * 16 + lr) * BSTRIDE;
#pragma unroll
                    for (int kk = 0; kk < 4; ++kk) {   // 4 x K=32 steps
                        const int ko = kk * 32 + half * 8;
                        v8bf blo = *reinterpret_cast<const v8bf*>(bcol + ko);
                        v8bf bhi = *reinterpret_cast<const v8bf*>(bcol + ko + 16);
                        v16bf bf = __builtin_shufflevector(
                            blo, bhi, 0, 1, 2, 3, 4, 5, 6, 7,
                            8, 9, 10, 11, 12, 13, 14, 15);
                        acc[t] = __builtin_amdgcn_wmma_f32_16x16x32_bf16(
                            false, afrag[kblk * 4 + kk], false, bf,
                            (short)0, acc[t], false, false);
                    }
                }
            }
        }

        if (valid) {
#pragma unroll
            for (int t = 0; t < 8; ++t) {
                const int col = mb * 128 + t * 16 + lr;
                const float bcol = bias[col];
#pragma unroll
                for (int r = 0; r < 8; ++r) {
                    const int orow = strip * 16 + half * 8 + r;
                    float v = acc[t][r] + bcol;
                    if (residual) v += residual[(size_t)orow * M + col];
                    if (relu) v = v > 0.f ? v : 0.f;
                    out[(size_t)orow * M + col] = v;
                }
            }
        }
    }
}

// ---------- stats init ----------------------------------------------------
__global__ void init_stats_kernel(unsigned* gmax, float* gsum) {
    int t = threadIdx.x;
    if (t < HEADS) { gmax[t] = 0x007FFFFFu /* f2ord(-inf) */; gsum[t] = 0.f; }
}

// ---------- edge scores + global per-head max -----------------------------
__global__ void scores_kernel(const float* __restrict__ Q, const float* __restrict__ Kf,
                              const int* __restrict__ src, const int* __restrict__ dst,
                              float* __restrict__ sc, unsigned* __restrict__ gmax) {
    __shared__ unsigned sm[HEADS];
    if (threadIdx.x < HEADS) sm[threadIdx.x] = 0x007FFFFFu;
    __syncthreads();
    const int e = blockIdx.x * blockDim.x + threadIdx.x;
    if (e < N_EDGES) {
        const int s = src[e], d = dst[e];
        const float4* qp = reinterpret_cast<const float4*>(Q  + (size_t)d * HIDDEN);
        const float4* kp = reinterpret_cast<const float4*>(Kf + (size_t)s * HIDDEN);
        float accs[HEADS] = {0.f, 0.f, 0.f, 0.f};
#pragma unroll
        for (int c = 0; c < 32; ++c) {
            float4 qv = qp[c], kv = kp[c];
            accs[c >> 3] += qv.x * kv.x + qv.y * kv.y + qv.z * kv.z + qv.w * kv.w;
        }
#pragma unroll
        for (int h = 0; h < HEADS; ++h) {
            float sv = accs[h] * 0.17677669529663687f;  // 1/sqrt(32)
            sc[(size_t)e * HEADS + h] = sv;
            atomicMax(&sm[h], f2ord(sv));
        }
    }
    __syncthreads();
    if (threadIdx.x < HEADS) atomicMax(&gmax[threadIdx.x], sm[threadIdx.x]);
}

// ---------- global per-head sum of exp(score - max) -----------------------
__global__ void sumexp_kernel(const float* __restrict__ sc,
                              const unsigned* __restrict__ gmax,
                              float* __restrict__ gsum) {
    __shared__ float ss[HEADS];
    if (threadIdx.x < HEADS) ss[threadIdx.x] = 0.f;
    __syncthreads();
    const int e = blockIdx.x * blockDim.x + threadIdx.x;
    if (e < N_EDGES) {
#pragma unroll
        for (int h = 0; h < HEADS; ++h) {
            float m = ord2f(gmax[h]);
            atomicAdd(&ss[h], expf(sc[(size_t)e * HEADS + h] - m));
        }
    }
    __syncthreads();
    if (threadIdx.x < HEADS) atomicAdd(&gsum[threadIdx.x], ss[threadIdx.x]);
}

// ---------- scatter: agg[dst] += v[src] * attn (f32 global atomics) -------
__global__ void scatter_kernel(const float* __restrict__ sc,
                               const unsigned* __restrict__ gmax,
                               const float* __restrict__ gsum,
                               const int* __restrict__ src, const int* __restrict__ dst,
                               const float* __restrict__ V, float* __restrict__ agg) {
    const int t = blockIdx.x * blockDim.x + threadIdx.x;
    if (t >= N_EDGES * HEADS) return;
    const int e = t >> 2, h = t & 3;
    const float m = ord2f(gmax[h]);
    const float w = expf(sc[(size_t)e * HEADS + h] - m) / gsum[h];
    const int s = src[e], d = dst[e];
    const float4* vp = reinterpret_cast<const float4*>(V + (size_t)s * HIDDEN + h * HEAD_DIM);
    float* ap = agg + (size_t)d * HIDDEN + h * HEAD_DIM;
#pragma unroll
    for (int c = 0; c < 8; ++c) {
        float4 vv = vp[c];
        atomicAdd(ap + c * 4 + 0, vv.x * w);
        atomicAdd(ap + c * 4 + 1, vv.y * w);
        atomicAdd(ap + c * 4 + 2, vv.z * w);
        atomicAdd(ap + c * 4 + 3, vv.w * w);
    }
}

// ==========================================================================
extern "C" void kernel_launch(void* const* d_in, const int* in_sizes, int n_in,
                              void* d_out, int out_size, void* d_ws, size_t ws_size,
                              hipStream_t stream) {
    (void)in_sizes; (void)n_in; (void)out_size; (void)ws_size;
    const float* x_in = (const float*)d_in[0];
    const int*   ei   = (const int*)d_in[1];     // [2, E] int32
    const float* Wi   = (const float*)d_in[2];
    const float* bi   = (const float*)d_in[3];
    const float* Wq   = (const float*)d_in[4];
    const float* bq   = (const float*)d_in[5];
    const float* Wk   = (const float*)d_in[6];
    const float* bk   = (const float*)d_in[7];
    const float* Wv   = (const float*)d_in[8];
    const float* bv   = (const float*)d_in[9];
    const float* Wo   = (const float*)d_in[10];
    const float* bo   = (const float*)d_in[11];
    const float* W1   = (const float*)d_in[12];
    const float* b1   = (const float*)d_in[13];
    const float* W2   = (const float*)d_in[14];
    const float* b2   = (const float*)d_in[15];
    const int* src = ei;
    const int* dst = ei + N_EDGES;

    const size_t NF = (size_t)N_NODES * HIDDEN;  // 6.4M floats
    float* ws  = (float*)d_ws;
    float* X   = ws;            // current x
    float* X2  = X  + NF;       // post-attention x; SC (scores) aliases this
    float* Qb  = X2 + NF;
    float* Kb  = Qb + NF;
    float* Vb  = Kb + NF;
    float* AG  = Vb + NF;
    float* HFF = Qb;            // FFN hidden [N,256] aliases Q+K (dead then)
    float* SC  = X2;            // scores [E,4] alias (dead before X2 written)
    unsigned* GMAX = (unsigned*)(AG + NF);
    float*    GSUM = (float*)(GMAX + HEADS);
    __bf16*   WT   = (__bf16*)(GSUM + 12);       // keep 32B alignment

    // bf16 transposed weight slots
    __bf16* WiT = WT;                                   // 128*128
    const size_t SQ = (size_t)HIDDEN * HIDDEN;          // 16384
    const size_t S1 = (size_t)HIDDEN * 2 * HIDDEN;      // 32768
    __bf16* LW[LAYERS][6];
    {
        __bf16* p = WiT + SQ;
        for (int l = 0; l < LAYERS; ++l) {
            LW[l][0] = p;            // WqT
            LW[l][1] = p + SQ;       // WkT
            LW[l][2] = p + 2 * SQ;   // WvT
            LW[l][3] = p + 3 * SQ;   // WoT
            LW[l][4] = p + 4 * SQ;            // W1T [256,128]
            LW[l][5] = p + 4 * SQ + S1;       // W2T [128,256]
            p += 4 * SQ + 2 * S1;
        }
    }

    // ---- weight prep (bf16 transpose) ----
    const int TB = 256;
    const int gSQ = (int)((SQ + TB - 1) / TB);
    const int gS1 = (int)((S1 + TB - 1) / TB);
    convT_kernel<<<gSQ, TB, 0, stream>>>(Wi, WiT, HIDDEN, HIDDEN);
    for (int l = 0; l < LAYERS; ++l) {
        convT_kernel<<<gSQ, TB, 0, stream>>>(Wq + l * SQ, LW[l][0], HIDDEN, HIDDEN);
        convT_kernel<<<gSQ, TB, 0, stream>>>(Wk + l * SQ, LW[l][1], HIDDEN, HIDDEN);
        convT_kernel<<<gSQ, TB, 0, stream>>>(Wv + l * SQ, LW[l][2], HIDDEN, HIDDEN);
        convT_kernel<<<gSQ, TB, 0, stream>>>(Wo + l * SQ, LW[l][3], HIDDEN, HIDDEN);
        convT_kernel<<<gS1, TB, 0, stream>>>(W1 + l * S1, LW[l][4], HIDDEN, 2 * HIDDEN);
        convT_kernel<<<gS1, TB, 0, stream>>>(W2 + l * S1, LW[l][5], 2 * HIDDEN, HIDDEN);
    }

    const int gemm_blocks = (N_NODES / 16 + 7) / 8;     // 391 (8 strips/block)
    auto gemm128 = [&](const float* A, const __bf16* Wt, const float* bias,
                       const float* res, float* out, int M, int relu) {
        gemm_bf16_wmma<128><<<gemm_blocks, 256, 0, stream>>>(A, Wt, bias, res,
                                                             out, N_NODES, M, relu);
    };
    auto gemm256 = [&](const float* A, const __bf16* Wt, const float* bias,
                       const float* res, float* out, int M, int relu) {
        gemm_bf16_wmma<256><<<gemm_blocks, 256, 0, stream>>>(A, Wt, bias, res,
                                                             out, N_NODES, M, relu);
    };

    // ---- input projection: X = x_in @ Wi + bi ----
    gemm128(x_in, WiT, bi, nullptr, X, HIDDEN, 0);

    const int gE  = (N_EDGES + TB - 1) / TB;            // 3125
    const int gEH = (N_EDGES * HEADS + TB - 1) / TB;    // 12500

    for (int l = 0; l < LAYERS; ++l) {
        // Q/K/V projections
        gemm128(X, LW[l][0], bq + l * HIDDEN, nullptr, Qb, HIDDEN, 0);
        gemm128(X, LW[l][1], bk + l * HIDDEN, nullptr, Kb, HIDDEN, 0);
        gemm128(X, LW[l][2], bv + l * HIDDEN, nullptr, Vb, HIDDEN, 0);

        // attention: global softmax over all edges per head
        init_stats_kernel<<<1, 32, 0, stream>>>(GMAX, GSUM);
        hipMemsetAsync(AG, 0, NF * sizeof(float), stream);
        scores_kernel<<<gE, TB, 0, stream>>>(Qb, Kb, src, dst, SC, GMAX);
        sumexp_kernel<<<gE, TB, 0, stream>>>(SC, GMAX, GSUM);
        scatter_kernel<<<gEH, TB, 0, stream>>>(SC, GMAX, GSUM, src, dst, Vb, AG);

        // out projection + residual: X2 = X + AG @ Wo + bo
        gemm128(AG, LW[l][3], bo + l * HIDDEN, X, X2, HIDDEN, 0);

        // FFN: HFF = relu(X2 @ W1 + b1);  x_next = X2 + HFF @ W2 + b2
        gemm128(X2, LW[l][4], b1 + l * 2 * HIDDEN, nullptr, HFF, 2 * HIDDEN, 1);
        float* xout = (l == LAYERS - 1) ? (float*)d_out : X;
        gemm256(HFF, LW[l][5], b2 + l * HIDDEN, X2, xout, HIDDEN, 0);
    }
}
